// Model_86423331930277
// MI455X (gfx1250) — compile-verified, matrix operand order checked by reading, not run on hardware
//
#include <hip/hip_runtime.h>
#include <hip/hip_bf16.h>

// ---------------------------------------------------------------------------
// LeNet + ResNet-fallback pipeline for MI455X (gfx1250, wave32, WMMA).
// Dominant cost: four 64->64 3x3 convs (~309 of ~315 GFLOP) -> implicit GEMM
// on v_wmma_f32_16x16x32_f16, f16 NHWC activations, fp32 accumulation.
// Block = 4 rows x 32 px x 64 ch (halo DRAM factor 1.5x); 8 waves =
// 4 N-tiles x 2 M-halves, 4 accumulators/wave (~96 VGPRs total).
// A-fragments for the 4 M-subtiles are batch-loaded into distinct registers
// so WMMAs wait on staggered dscnt thresholds instead of full drains.
// Halo staged to LDS with gfx1250 GLOBAL_LOAD_ASYNC_TO_LDS_B128 (ASYNCcnt).
// ---------------------------------------------------------------------------

typedef __attribute__((ext_vector_type(16))) _Float16 v16h;
typedef __attribute__((ext_vector_type(8)))  float    v8f;

#define B_TOTAL 1024
#define NCLS    10
#define CHUNK   128          // resnet batch chunk (3 rotating 16MiB f16 buffers)
#define USE_ASYNC_STAGE 1    // gfx1250 GLOBAL_LOAD_ASYNC_TO_LDS_B128 staging

// ---------------------------------------------------------------- utilities
__global__ void zero_f32(float* p, int n) {
    int i = blockIdx.x * blockDim.x + threadIdx.x;
    if (i < n) p[i] = 0.f;
}

// ------------------------------------------------------------------- LeNet
// conv1 (6,3,5,5) VALID + relu + 2x2 maxpool : x[B,3,32,32] -> pool1[B,6,14,14]
__global__ void lenet_c1(const float* __restrict__ x, const float* __restrict__ w,
                         const float* __restrict__ b, float* __restrict__ pool1) {
    int id = blockIdx.x * blockDim.x + threadIdx.x;
    if (id >= B_TOTAL * 6 * 14 * 14) return;
    int bi = id / (6 * 196);
    int r  = id % (6 * 196);
    int oc = r / 196;
    int pp = r % 196;
    int py = pp / 14, px = pp % 14;
    float mx = 0.f;
    for (int dy = 0; dy < 2; ++dy)
        for (int dx = 0; dx < 2; ++dx) {
            int oy = 2 * py + dy, ox = 2 * px + dx;
            float s = b[oc];
            for (int ci = 0; ci < 3; ++ci)
                for (int kh = 0; kh < 5; ++kh)
                    for (int kw = 0; kw < 5; ++kw)
                        s += x[((bi * 3 + ci) * 32 + oy + kh) * 32 + ox + kw] *
                             w[((oc * 3 + ci) * 5 + kh) * 5 + kw];
            s = fmaxf(s, 0.f);
            mx = fmaxf(mx, s);
        }
    pool1[((bi * 6 + oc) * 14 + py) * 14 + px] = mx;
}

// conv2 (16,6,5,5) VALID + relu + pool : pool1 -> feat[B,400] (c*25+y*5+x)
__global__ void lenet_c2(const float* __restrict__ p1, const float* __restrict__ w,
                         const float* __restrict__ b, float* __restrict__ feat) {
    int id = blockIdx.x * blockDim.x + threadIdx.x;
    if (id >= B_TOTAL * 16 * 25) return;
    int bi = id / 400;
    int r  = id % 400;
    int oc = r / 25;
    int pp = r % 25;
    int py = pp / 5, px = pp % 5;
    float mx = 0.f;
    for (int dy = 0; dy < 2; ++dy)
        for (int dx = 0; dx < 2; ++dx) {
            int oy = 2 * py + dy, ox = 2 * px + dx;
            float s = b[oc];
            for (int ci = 0; ci < 6; ++ci)
                for (int kh = 0; kh < 5; ++kh)
                    for (int kw = 0; kw < 5; ++kw)
                        s += p1[((bi * 6 + ci) * 14 + oy + kh) * 14 + ox + kw] *
                             w[((oc * 6 + ci) * 5 + kh) * 5 + kw];
            s = fmaxf(s, 0.f);
            mx = fmaxf(mx, s);
        }
    feat[bi * 400 + oc * 25 + py * 5 + px] = mx;
}

// fc1(400->120) relu, fc2(120->84) relu, fc3(84->10); one 128-thread block/img
__global__ __launch_bounds__(128)
void lenet_fc(const float* __restrict__ feat,
              const float* __restrict__ w1, const float* __restrict__ b1,
              const float* __restrict__ w2, const float* __restrict__ b2,
              const float* __restrict__ w3, const float* __restrict__ b3,
              float* __restrict__ lenet_out) {
    __shared__ float sh[400 + 120 + 84];
    int bi = blockIdx.x, t = threadIdx.x;
    for (int i = t; i < 400; i += 128) sh[i] = feat[bi * 400 + i];
    __syncthreads();
    if (t < 120) {
        float s = b1[t];
        for (int i = 0; i < 400; ++i) s += sh[i] * w1[i * 120 + t];
        sh[400 + t] = fmaxf(s, 0.f);
    }
    __syncthreads();
    if (t < 84) {
        float s = b2[t];
        for (int i = 0; i < 120; ++i) s += sh[400 + i] * w2[i * 84 + t];
        sh[520 + t] = fmaxf(s, 0.f);
    }
    __syncthreads();
    if (t < NCLS) {
        float s = b3[t];
        for (int i = 0; i < 84; ++i) s += sh[520 + i] * w3[i * NCLS + t];
        lenet_out[bi * NCLS + t] = s;
    }
}

// argmax / softmax top-2 margin / confusion-matrix scatter-add
__global__ void classify(const float* __restrict__ lenet_out,
                         const float* __restrict__ thr, const int* __restrict__ labels,
                         int* __restrict__ pred, int* __restrict__ lowc,
                         float* __restrict__ conf) {
    int bi = blockIdx.x * blockDim.x + threadIdx.x;
    if (bi >= B_TOTAL) return;
    float v[NCLS];
    for (int c = 0; c < NCLS; ++c) v[c] = lenet_out[bi * NCLS + c];
    int pm = 0; float mx = v[0];
    for (int c = 1; c < NCLS; ++c) if (v[c] > mx) { mx = v[c]; pm = c; }
    float s = 0.f, e[NCLS];
    for (int c = 0; c < NCLS; ++c) { e[c] = __expf(v[c] - mx); s += e[c]; }
    float p1 = -1.f, p2 = -1.f;
    for (int c = 0; c < NCLS; ++c) {
        float p = e[c] / s;
        if (p > p1) { p2 = p1; p1 = p; } else if (p > p2) p2 = p;
    }
    pred[bi] = pm;
    lowc[bi] = (p1 - p2) <= thr[0] ? 1 : 0;
    atomicAdd(&conf[labels[bi] * NCLS + pm], 1.0f);
}

// ------------------------------------------------------------------ ResNet
// conv0: x[NCHW fp32, 3ch] -> h0[NHWC f16, 64ch] + relu (K=27, negligible cost)
__global__ __launch_bounds__(64)
void resnet_conv0(const float* __restrict__ x, const float* __restrict__ w,
                  const float* __restrict__ b, _Float16* __restrict__ out, int c0) {
    int pix = blockIdx.x;               // img_local*1024 + py*32 + px
    int img = pix >> 10;
    int py  = (pix >> 5) & 31, px = pix & 31;
    int co  = threadIdx.x;
    const float* xb = x + (size_t)(c0 + img) * 3 * 1024;
    float s = b[co];
    for (int ci = 0; ci < 3; ++ci)
        for (int kh = 0; kh < 3; ++kh) {
            int iy = py + kh - 1;
            if (iy < 0 || iy > 31) continue;
            for (int kw = 0; kw < 3; ++kw) {
                int ix = px + kw - 1;
                if (ix < 0 || ix > 31) continue;
                s += xb[(ci * 32 + iy) * 32 + ix] * w[((co * 3 + ci) * 3 + kh) * 3 + kw];
            }
        }
    out[(size_t)pix * 64 + co] = (_Float16)fmaxf(s, 0.f);
}

// Pack OIHW fp32 [64,64,3,3] weights into WMMA B-fragment order (f16):
// fragment f = ntile(4) * 18 + kstep; per the ISA 16-bit operand lane layout:
// lane L, half t: j=t/2,p=t&1; k=(j<4 ? (L<16?0:8)+2j : (L<16?16:24)+2(j-4))+p
// kstep kk -> tap=kk/2 (kh,kw), cbase=(kk&1)*32; ci=cbase+k; co=ntile*16+(L&15)
__global__ void pack_w3x3(const float* __restrict__ w, _Float16* __restrict__ dst) {
    int id = blockIdx.x * blockDim.x + threadIdx.x;      // 4*18*32*16 = 36864
    if (id >= 4 * 18 * 512) return;
    int frag = id >> 9;
    int lane = (id >> 4) & 31;
    int t    = id & 15;
    int nt = frag / 18, kk = frag % 18;
    int tap = kk >> 1, cb = (kk & 1) << 5;
    int kh = tap / 3, kw = tap % 3;
    int hi = (lane >> 4) & 1;
    int j = t >> 1, p = t & 1;
    int kloc = (j < 4) ? ((hi ? 8 : 0) + 2 * j + p)
                       : ((hi ? 24 : 16) + 2 * (j - 4) + p);
    int ci = cb + kloc;
    int co = nt * 16 + (lane & 15);
    dst[id] = (_Float16)w[((co * 64 + ci) * 3 + kh) * 3 + kw];
}

// 3x3 SAME conv, 64->64 ch, implicit GEMM on v_wmma_f32_16x16x32_f16.
// Block = 256 threads = 8 waves: wave w -> N-tile (w&3), M-half (w>>2).
// Block output: 4 rows x 32 px x 64 ch; each wave: 2 rows x 32 px x 16 ch
// = 4 accumulators, 72 WMMAs. K = 9 taps x 64 ch = 18 steps of 32.
// Per k-step: batch-load 4 A-fragments (8 ds_load_b128 clause) then issue
// 4 WMMAs -> staggered s_wait_dscnt instead of per-WMMA full drains.
// Halo tile (6 rows x 34 cols x 64ch f16 = 25.5KB) staged into LDS with
// GLOBAL_LOAD_ASYNC_TO_LDS_B128 (ASYNCcnt), zero-padded borders via ds stores.
// Epilogue fuses bias + optional residual + relu + f16 convert.
__global__ __launch_bounds__(256)
void resconv3x3_wmma(const _Float16* __restrict__ in, const _Float16* __restrict__ wp,
                     const float* __restrict__ bias, const _Float16* __restrict__ res,
                     _Float16* __restrict__ out) {
    __shared__ __align__(16) _Float16 tile[6 * 34 * 64];   // 26112 B
    int bid = blockIdx.x;
    int img = bid >> 3;
    int y0  = (bid & 7) << 2;           // row group: 4 output rows
    int t   = threadIdx.x;

    // Stage halo: 6 rows x 34 cols, each column = 64 f16 channels (128 B).
    if (t < 6 * 34) {
        int r = t / 34, c = t % 34;
        int gy = y0 + r - 1, gx = c - 1;
        _Float16* dst = &tile[(r * 34 + c) * 64];
        if (gy >= 0 && gy < 32 && gx >= 0 && gx < 32) {
            const _Float16* src = &in[((size_t)(img * 32 + gy) * 32 + gx) * 64];
#if USE_ASYNC_STAGE
            unsigned la = (unsigned)(size_t)dst;             // LDS byte address
            unsigned long long ga = (unsigned long long)(size_t)src;
            #pragma unroll
            for (int i = 0; i < 8; ++i)
                asm volatile("global_load_async_to_lds_b128 %0, %1, off"
                             :: "v"(la + 16u * i), "v"(ga + 16ull * i) : "memory");
#else
            #pragma unroll
            for (int i = 0; i < 8; ++i)
                ((float4*)dst)[i] = ((const float4*)src)[i];
#endif
        } else {
            float4 z = make_float4(0.f, 0.f, 0.f, 0.f);
            #pragma unroll
            for (int i = 0; i < 8; ++i) ((float4*)dst)[i] = z;
        }
    }
#if USE_ASYNC_STAGE
    asm volatile("s_wait_asynccnt 0" ::: "memory");
#endif
    __syncthreads();

    int lane = t & 31;
    int wv   = t >> 5;                  // wave id 0..7
    int nt   = wv & 3;                  // output-channel tile
    int mh   = wv >> 2;                 // M-half: rows {0,1} or {2,3}
    int hi   = (lane >> 4) & 1;
    int m    = lane & 15;               // A: pixel within M-subtile; D: channel
    v8f acc[4] = {};

    // K loop: 9 taps x 2 channel-halves; unroll window capped to bound VGPRs.
    #pragma unroll 3
    for (int tap = 0; tap < 9; ++tap) {
        int kh = tap / 3, kw = tap % 3;
        #pragma unroll
        for (int half = 0; half < 2; ++half) {
            int kk = tap * 2 + half;
            int cb = half << 5;
            // B fragment: pre-packed fragment-major -> 32 contiguous B per lane
            const _Float16* bp = wp + (((size_t)nt * 18 + kk) * 32 + lane) * 16;
            v16h b;
            ((float4*)&b)[0] = *(const float4*)(bp);
            ((float4*)&b)[1] = *(const float4*)(bp + 8);
            // Batch-load all 4 A fragments into distinct registers first...
            v16h a[4];
            #pragma unroll
            for (int s = 0; s < 4; ++s) {
                int row = (mh << 1) + (s >> 1) + kh;
                int col = ((s & 1) << 4) + m + kw;
                const _Float16* ap = &tile[(row * 34 + col) * 64 + cb];
                ((float4*)&a[s])[0] = *(const float4*)(ap + (hi ? 8 : 0));
                ((float4*)&a[s])[1] = *(const float4*)(ap + 16 + (hi ? 8 : 0));
            }
            // ...then issue the 4 WMMAs back-to-back (staggered dscnt waits).
            #pragma unroll
            for (int s = 0; s < 4; ++s)
                acc[s] = __builtin_amdgcn_wmma_f32_16x16x32_f16(
                             false, a[s], false, b, (short)0, acc[s], false, false);
        }
    }

    // C/D layout: reg r, lanes 0-15 -> M=r, lanes 16-31 -> M=r+8; N = lane&15
    int ch = (nt << 4) + m;
    float bch = bias[ch];
    #pragma unroll
    for (int s = 0; s < 4; ++s) {
        int y  = y0 + (mh << 1) + (s >> 1);
        int xb = (s & 1) << 4;
        #pragma unroll
        for (int r = 0; r < 8; ++r) {
            int px = xb + r + (hi ? 8 : 0);
            size_t oidx = ((size_t)(img * 32 + y) * 32 + px) * 64 + ch;
            float v = acc[s][r] + bch;
            if (res) v += (float)res[oidx];
            out[oidx] = (_Float16)fmaxf(v, 0.f);
        }
    }
}

// global average pool over 32x32 + fc 64->10
__global__ __launch_bounds__(256)
void resnet_poolfc(const _Float16* __restrict__ h, const float* __restrict__ rfc,
                   const float* __restrict__ rfb, float* __restrict__ res_out, int c0) {
    __shared__ float part[256];
    __shared__ float mean[64];
    int img = blockIdx.x, t = threadIdx.x;
    int ch = t & 63, g = t >> 6;
    const _Float16* hb = h + (size_t)img * 1024 * 64;
    float s = 0.f;
    for (int p = g * 256; p < g * 256 + 256; ++p) s += (float)hb[(size_t)p * 64 + ch];
    part[t] = s;
    __syncthreads();
    if (t < 64)
        mean[t] = (part[t] + part[t + 64] + part[t + 128] + part[t + 192]) * (1.0f / 1024.0f);
    __syncthreads();
    if (t < NCLS) {
        float o = rfb[t];
        for (int c = 0; c < 64; ++c) o += mean[c] * rfc[c * NCLS + t];
        res_out[(size_t)(c0 + img) * NCLS + t] = o;
    }
}

// masked merge: out = low_conf ? resnet : lenet
__global__ void merge_out(const float* __restrict__ lenet_out, const float* __restrict__ res_out,
                          const int* __restrict__ lowc, float* __restrict__ out) {
    int i = blockIdx.x * blockDim.x + threadIdx.x;
    if (i >= B_TOTAL * NCLS) return;
    out[i] = lowc[i / NCLS] ? res_out[i] : lenet_out[i];
}

// ---------------------------------------------------------------------------
extern "C" void kernel_launch(void* const* d_in, const int* in_sizes, int n_in,
                              void* d_out, int out_size, void* d_ws, size_t ws_size,
                              hipStream_t stream) {
    (void)in_sizes; (void)n_in; (void)out_size; (void)ws_size;
    const float* x    = (const float*)d_in[0];
    const float* thr  = (const float*)d_in[1];
    const int*   lab  = (const int*)d_in[2];
    const float* lw1  = (const float*)d_in[3];
    const float* lb1  = (const float*)d_in[4];
    const float* lw2  = (const float*)d_in[5];
    const float* lb2  = (const float*)d_in[6];
    const float* lfc1 = (const float*)d_in[7];
    const float* lfb1 = (const float*)d_in[8];
    const float* lfc2 = (const float*)d_in[9];
    const float* lfb2 = (const float*)d_in[10];
    const float* lfc3 = (const float*)d_in[11];
    const float* lfb3 = (const float*)d_in[12];
    const float* rw0  = (const float*)d_in[13];
    const float* rb0  = (const float*)d_in[14];
    const float* rw1a = (const float*)d_in[15];
    const float* rb1a = (const float*)d_in[16];
    const float* rw1b = (const float*)d_in[17];
    const float* rb1b = (const float*)d_in[18];
    const float* rw2a = (const float*)d_in[19];
    const float* rb2a = (const float*)d_in[20];
    const float* rw2b = (const float*)d_in[21];
    const float* rb2b = (const float*)d_in[22];
    const float* rfc  = (const float*)d_in[23];
    const float* rfb  = (const float*)d_in[24];

    float* out  = (float*)d_out;            // [B*10] logits, then [10*10] conf
    float* conf = out + B_TOTAL * NCLS;

    // workspace carve-up (all offsets 256B-aligned)
    char* w = (char*)d_ws;
    size_t off = 0;
    float* pool1      = (float*)(w + off); off += (size_t)B_TOTAL * 6 * 14 * 14 * 4;
    float* feat       = (float*)(w + off); off += (size_t)B_TOTAL * 400 * 4;
    float* lenet_out  = (float*)(w + off); off += (size_t)B_TOTAL * NCLS * 4;
    float* res_out    = (float*)(w + off); off += (size_t)B_TOTAL * NCLS * 4;
    int*   pred       = (int*)(w + off);   off += (size_t)B_TOTAL * 4;
    int*   lowc       = (int*)(w + off);   off += (size_t)B_TOTAL * 4;
    _Float16* wpack   = (_Float16*)(w + off); off += (size_t)4 * 36864 * 2; // 4 layers
    size_t xbytes = (size_t)CHUNK * 1024 * 64 * 2;                          // 16 MiB
    _Float16* X0 = (_Float16*)(w + off); off += xbytes;
    _Float16* X1 = (_Float16*)(w + off); off += xbytes;
    _Float16* X2 = (_Float16*)(w + off); off += xbytes;

    // --- LeNet path (everything in fp32, <2% of total FLOPs) ---
    zero_f32<<<1, 128, 0, stream>>>(conf, NCLS * NCLS);
    lenet_c1<<<(B_TOTAL * 6 * 196 + 255) / 256, 256, 0, stream>>>(x, lw1, lb1, pool1);
    lenet_c2<<<(B_TOTAL * 16 * 25 + 255) / 256, 256, 0, stream>>>(pool1, lw2, lb2, feat);
    lenet_fc<<<B_TOTAL, 128, 0, stream>>>(feat, lfc1, lfb1, lfc2, lfb2, lfc3, lfb3, lenet_out);
    classify<<<(B_TOTAL + 255) / 256, 256, 0, stream>>>(lenet_out, thr, lab, pred, lowc, conf);

    // --- pack resnet 3x3 weights into WMMA B-fragment order (f16) ---
    pack_w3x3<<<144, 256, 0, stream>>>(rw1a, wpack + 0 * 36864);
    pack_w3x3<<<144, 256, 0, stream>>>(rw1b, wpack + 1 * 36864);
    pack_w3x3<<<144, 256, 0, stream>>>(rw2a, wpack + 2 * 36864);
    pack_w3x3<<<144, 256, 0, stream>>>(rw2b, wpack + 3 * 36864);

    // --- ResNet path, chunked over batch, 3 rotating f16 NHWC buffers ---
    int convBlocks = CHUNK * 8;             // 8 row-groups (4 rows each) per image
    for (int c0 = 0; c0 < B_TOTAL; c0 += CHUNK) {
        resnet_conv0<<<CHUNK * 1024, 64, 0, stream>>>(x, rw0, rb0, X0, c0);
        // h0=X0; r=conv1a(X0)->X1; h1=relu(X0+conv1b(X1))->X2
        resconv3x3_wmma<<<convBlocks, 256, 0, stream>>>(X0, wpack + 0 * 36864, rb1a, nullptr, X1);
        resconv3x3_wmma<<<convBlocks, 256, 0, stream>>>(X1, wpack + 1 * 36864, rb1b, X0, X2);
        // r=conv2a(X2)->X1; h2=relu(X2+conv2b(X1))->X0
        resconv3x3_wmma<<<convBlocks, 256, 0, stream>>>(X2, wpack + 2 * 36864, rb2a, nullptr, X1);
        resconv3x3_wmma<<<convBlocks, 256, 0, stream>>>(X1, wpack + 3 * 36864, rb2b, X2, X0);
        resnet_poolfc<<<CHUNK, 256, 0, stream>>>(X0, rfc, rfb, res_out, c0);
    }

    merge_out<<<(B_TOTAL * NCLS + 255) / 256, 256, 0, stream>>>(lenet_out, res_out, lowc, out);
}
